// Enhanced_Physics_Attention_with_Sonata_24953759990239
// MI455X (gfx1250) — compile-verified, matrix operand order checked by reading, not run on hardware
//
#include <hip/hip_runtime.h>
#include <hip/hip_bf16.h>

// ---------------------------------------------------------------------------
// CDNA5 (gfx1250) implementation: all contractions via v_wmma_f32_16x16x32_bf16
// (wave32, 16x16 C tile / 8 VGPRs per lane). Context attention is fused
// flash-style so the (B,H,G,M) score tensor is never materialized.
// GEMM: one wave owns a 64x16 output strip (4 WMMA tiles sharing one B
// fragment per k-step), double-buffered fragment loads, ballot-based batch
// masking for the token-aggregation contraction.
// ---------------------------------------------------------------------------

typedef __bf16 bf16_t;
typedef __attribute__((ext_vector_type(16))) __bf16 v16bf;
typedef __attribute__((ext_vector_type(8)))  float  v8f;

#define N_TOK 16384
#define DIMX  512
#define HEADS 8
#define DHEAD 64
#define GSLOT 64
#define BATCH 8
#define MSON  4096
#define SCALE 0.125f          // 64^-0.5
#define NEGBIG (-3.0e38f)

union FragU { v16bf v; uint4 q[2]; };

// A-operand (and transposed-B) fragment, contiguous along K.
// ISA 7.12.2 16-bit A 16x32: lanes 0-15 hold K {k0..k0+7, k0+16..k0+23},
// lanes 16-31 hold K {k0+8..k0+15, k0+24..k0+31}.
__device__ inline v16bf frag_contig(const bf16_t* base, int k0, int half) {
  FragU f;
  f.q[0] = *reinterpret_cast<const uint4*>(base + k0 + half * 8);
  f.q[1] = *reinterpret_cast<const uint4*>(base + k0 + 16 + half * 8);
  return f.v;
}

// Strided fragment (row-major B, or transposed A); branch-free per-K masking
// from a wave-uniform ballot word (bit kk == keep element at k0+kk).
template<int MASKED>
__device__ inline v16bf frag_strided(const bf16_t* p, long long ld, int idx,
                                     int k0, int half, unsigned mbits) {
  v16bf v;
#pragma unroll
  for (int i = 0; i < 16; ++i) {
    int kk = ((i >> 3) << 4) + half * 8 + (i & 7);   // local k in [0,32)
    bf16_t val = p[(long long)(k0 + kk) * ld + idx];
    if (MASKED) val = ((mbits >> kk) & 1u) ? val : (bf16_t)0.0f;
    v[i] = val;
  }
  return v;
}

// ---------------------------------------------------------------------------
// Generic batched WMMA GEMM: C[z] = alpha * opA(A[z]) @ opB(B[z]) + bias
// One wave computes a 64x16 strip (4 vertical 16x16 tiles, shared B fragment).
// z1 = z/zdiv, z2 = z%zdiv select per-slice offsets.
// ---------------------------------------------------------------------------
template<int TA, int TB, int MASKED>
__global__ __launch_bounds__(32)
void gemm_wmma(const bf16_t* __restrict__ A, long long lda,
               const bf16_t* __restrict__ Bm, long long ldb,
               const float* __restrict__ bias, float alpha,
               float* __restrict__ Cf, bf16_t* __restrict__ Cb, long long ldc,
               int K, int zdiv,
               long long aO1, long long aO2, long long bO1, long long bO2,
               long long cO1, long long cO2,
               const int* __restrict__ kmask, int maskDiv) {
  int z  = blockIdx.z;
  int z1 = z / zdiv, z2 = z % zdiv;
  const bf16_t* Ab = A  + z1 * aO1 + z2 * aO2;
  const bf16_t* Bb = Bm + z1 * bO1 + z2 * bO2;
  long long cOff = z1 * cO1 + z2 * cO2;
  int maskVal = MASKED ? (z / maskDiv) : 0;

  int lane = threadIdx.x & 31;
  int lr   = lane & 15;
  int half = lane >> 4;
  int m0 = blockIdx.x * 64;
  int n0 = blockIdx.y * 16;

  const bf16_t* aRow[4];
#pragma unroll
  for (int t = 0; t < 4; ++t)
    aRow[t] = Ab + (long long)(m0 + t * 16 + lr) * lda;     // contig-A bases
  const bf16_t* bCol = Bb + (long long)(n0 + lr) * ldb;     // transB base

  auto ballotMask = [&](int k0) -> unsigned {
    if (!MASKED) return 0u;
    return (unsigned)__ballot(kmask[k0 + lane] == maskVal); // wave32: low 32b
  };
  auto loadA = [&](int t, int k0, unsigned mb) -> v16bf {
    if (TA) return frag_strided<MASKED>(Ab, lda, m0 + t * 16 + lr, k0, half, mb);
    __builtin_prefetch(aRow[t] + k0 + 64, 0, 1);
    return frag_contig(aRow[t], k0, half);
  };
  auto loadB = [&](int k0) -> v16bf {
    if (TB) return frag_contig(bCol, k0, half);
    return frag_strided<0>(Bb, ldb, n0 + lr, k0, half, 0u);
  };

  v8f acc[4];
#pragma unroll
  for (int t = 0; t < 4; ++t)
#pragma unroll
    for (int r = 0; r < 8; ++r) acc[t][r] = 0.0f;

  // software-pipelined main loop: next k-step's fragments load under 4 WMMAs
  unsigned mb = ballotMask(0);
  v16bf bcur = loadB(0);
  v16bf a0 = loadA(0, 0, mb), a1 = loadA(1, 0, mb),
        a2 = loadA(2, 0, mb), a3 = loadA(3, 0, mb);
  for (int k0 = 0; k0 < K; k0 += 32) {
    int kn = k0 + 32;
    v16bf bn = bcur, n0f = a0, n1f = a1, n2f = a2, n3f = a3;
    unsigned mbn = 0u;
    if (kn < K) {
      mbn = ballotMask(kn);
      bn  = loadB(kn);
      n0f = loadA(0, kn, mbn); n1f = loadA(1, kn, mbn);
      n2f = loadA(2, kn, mbn); n3f = loadA(3, kn, mbn);
    }
    acc[0] = __builtin_amdgcn_wmma_f32_16x16x32_bf16(false, a0, false, bcur,
                                                     (short)0, acc[0], false, false);
    acc[1] = __builtin_amdgcn_wmma_f32_16x16x32_bf16(false, a1, false, bcur,
                                                     (short)0, acc[1], false, false);
    acc[2] = __builtin_amdgcn_wmma_f32_16x16x32_bf16(false, a2, false, bcur,
                                                     (short)0, acc[2], false, false);
    acc[3] = __builtin_amdgcn_wmma_f32_16x16x32_bf16(false, a3, false, bcur,
                                                     (short)0, acc[3], false, false);
    bcur = bn; a0 = n0f; a1 = n1f; a2 = n2f; a3 = n3f; mb = mbn;
  }

  int col = n0 + lr;
  float bv = bias ? bias[col] : 0.0f;
#pragma unroll
  for (int t = 0; t < 4; ++t) {
#pragma unroll
    for (int r = 0; r < 8; ++r) {
      int row = m0 + t * 16 + r + 8 * half;   // C layout: row = reg + 8*half
      float val = alpha * acc[t][r] + bv;
      long long off = cOff + (long long)row * ldc + col;
      if (Cf) Cf[off] = val;
      if (Cb) Cb[off] = (bf16_t)val;
    }
  }
}

// ---------------------------------------------------------------------------
// f32 -> bf16 convert (grid-stride)
// ---------------------------------------------------------------------------
__global__ void f32_to_bf16(const float* __restrict__ s, bf16_t* __restrict__ d,
                            long long n) {
  long long i = (long long)blockIdx.x * blockDim.x + threadIdx.x;
  long long stride = (long long)gridDim.x * blockDim.x;
  for (; i < n; i += stride) d[i] = (bf16_t)s[i];
}

__global__ void zero_f32(float* p, int n) {
  int i = blockIdx.x * blockDim.x + threadIdx.x;
  if (i < n) p[i] = 0.0f;
}

// ---------------------------------------------------------------------------
// Row softmax over rows of length 64 (one wave per row). Optional 1/temp[row%mod]
// pre-scale (slice softmax). Writes f32 in place + bf16 copy.
// ---------------------------------------------------------------------------
__global__ __launch_bounds__(256)
void softmax_rows64(float* __restrict__ X, bf16_t* __restrict__ Ob, int nrows,
                    const float* __restrict__ temp, int tempMod) {
  int wave = threadIdx.x >> 5, lane = threadIdx.x & 31;
  int row = blockIdx.x * 8 + wave;
  if (row >= nrows) return;
  float* x = X + (long long)row * 64;
  float s = temp ? (1.0f / temp[row % tempMod]) : 1.0f;
  float v0 = x[lane] * s, v1 = x[lane + 32] * s;
  float m = fmaxf(v0, v1);
#pragma unroll
  for (int d = 1; d < 32; d <<= 1) m = fmaxf(m, __shfl_xor(m, d, 32));
  float e0 = __expf(v0 - m), e1 = __expf(v1 - m);
  float sum = e0 + e1;
#pragma unroll
  for (int d = 1; d < 32; d <<= 1) sum += __shfl_xor(sum, d, 32);
  float inv = 1.0f / sum;
  e0 *= inv; e1 *= inv;
  x[lane] = e0; x[lane + 32] = e1;
  if (Ob) {
    Ob[(long long)row * 64 + lane]      = (bf16_t)e0;
    Ob[(long long)row * 64 + lane + 32] = (bf16_t)e1;
  }
}

// ---------------------------------------------------------------------------
// norms[b,h,g] = sum_n (bi[n]==b) * w[n,h,g]   (partial chunks + atomicAdd)
// ---------------------------------------------------------------------------
__global__ __launch_bounds__(64)
void norms_partial(const float* __restrict__ w, const int* __restrict__ bi,
                   float* __restrict__ norms, int chunk) {
  int z = blockIdx.x;               // b*H + h
  int b = z >> 3, h = z & 7;
  int g = threadIdx.x;
  int n0 = blockIdx.y * chunk;
  int n1 = n0 + chunk; if (n1 > N_TOK) n1 = N_TOK;
  float acc = 0.0f;
  for (int n = n0; n < n1; ++n)
    if (bi[n] == b) acc += w[(long long)n * 512 + h * 64 + g];
  atomicAdd(&norms[z * 64 + g], acc);
}

__global__ void tokens_finalize(const float* __restrict__ tok,
                                const float* __restrict__ norms,
                                bf16_t* __restrict__ tokb, int n) {
  int i = blockIdx.x * blockDim.x + threadIdx.x;
  if (i < n) tokb[i] = (bf16_t)(tok[i] / (norms[i >> 6] + 1e-5f));
}

// ---------------------------------------------------------------------------
// Fused context attention (flash-style):
//   out_tok[b,h,g,:] += softmax_m(mask(out_tok @ ks^T * SCALE)) @ vs
// One wave per (z = b*H+h, 16-row g tile). Online softmax over M=4096 keys
// in 32-key chunks; P tile goes through LDS to become a WMMA A-operand.
// ---------------------------------------------------------------------------
__global__ __launch_bounds__(32)
void ctx_attn(const bf16_t* __restrict__ otb, const bf16_t* __restrict__ ksb,
              const bf16_t* __restrict__ vsb, const int* __restrict__ sbi,
              float* __restrict__ out_tok) {
  __shared__ __align__(16) bf16_t ldsP[16 * 32];
  int z = blockIdx.y;
  int b = z >> 3, h = z & 7;
  int g0 = blockIdx.x * 16;
  int lane = threadIdx.x, lr = lane & 15, half = lane >> 4;

  const bf16_t* qbase = otb + (long long)z * 4096 + (long long)(g0 + lr) * 64;
  v16bf qa0 = frag_contig(qbase, 0, half);
  v16bf qa1 = frag_contig(qbase, 32, half);

  v8f o[4];
  float mst[8], lst[8];
#pragma unroll
  for (int r = 0; r < 8; ++r) {
    mst[r] = NEGBIG; lst[r] = 0.0f;
    o[0][r] = o[1][r] = o[2][r] = o[3][r] = 0.0f;
  }

  for (int m0 = 0; m0 < MSON; m0 += 32) {
    // ---- S = Q @ K^T for two 16-key subtiles -------------------------------
    v8f s0, s1;
#pragma unroll
    for (int r = 0; r < 8; ++r) { s0[r] = 0.0f; s1[r] = 0.0f; }
    {
      const bf16_t* kb0 = ksb + (long long)(m0 + lr) * 512 + h * 64;
      const bf16_t* kb1 = ksb + (long long)(m0 + 16 + lr) * 512 + h * 64;
      v16bf k00 = frag_contig(kb0, 0, half), k01 = frag_contig(kb0, 32, half);
      v16bf k10 = frag_contig(kb1, 0, half), k11 = frag_contig(kb1, 32, half);
      s0 = __builtin_amdgcn_wmma_f32_16x16x32_bf16(false, qa0, false, k00,
                                                   (short)0, s0, false, false);
      s0 = __builtin_amdgcn_wmma_f32_16x16x32_bf16(false, qa1, false, k01,
                                                   (short)0, s0, false, false);
      s1 = __builtin_amdgcn_wmma_f32_16x16x32_bf16(false, qa0, false, k10,
                                                   (short)0, s1, false, false);
      s1 = __builtin_amdgcn_wmma_f32_16x16x32_bf16(false, qa1, false, k11,
                                                   (short)0, s1, false, false);
    }
    bool ok0 = (sbi[m0 + lr] == b);
    bool ok1 = (sbi[m0 + 16 + lr] == b);

    // ---- online softmax (per-row stats replicated over each 16-lane half) --
#pragma unroll
    for (int r = 0; r < 8; ++r) {
      float f0 = ok0 ? s0[r] * SCALE : NEGBIG;
      float f1 = ok1 ? s1[r] * SCALE : NEGBIG;
      float rm = fmaxf(f0, f1);
#pragma unroll
      for (int d = 1; d < 16; d <<= 1) rm = fmaxf(rm, __shfl_xor(rm, d, 16));
      float nm = fmaxf(mst[r], rm);
      float corr = __expf(mst[r] - nm);
      float e0 = ok0 ? __expf(f0 - nm) : 0.0f;
      float e1 = ok1 ? __expf(f1 - nm) : 0.0f;
      float rs = e0 + e1;
#pragma unroll
      for (int d = 1; d < 16; d <<= 1) rs += __shfl_xor(rs, d, 16);
      lst[r] = lst[r] * corr + rs;
      mst[r] = nm;
      o[0][r] *= corr; o[1][r] *= corr; o[2][r] *= corr; o[3][r] *= corr;
      int row = r + 8 * half;                    // C-layout -> LDS (row, key)
      ldsP[row * 32 + lr]      = (bf16_t)e0;
      ldsP[row * 32 + 16 + lr] = (bf16_t)e1;
    }
    __syncthreads();

    // ---- O += P @ V (K=32) -------------------------------------------------
    v16bf pa = frag_contig(ldsP + lr * 32, 0, half);   // ds_load_b128 x2
#pragma unroll
    for (int tt = 0; tt < 4; ++tt) {
      v16bf vb = frag_strided<0>(vsb + (long long)m0 * 512 + h * 64 + tt * 16,
                                 512, lr, 0, half, 0u);
      o[tt] = __builtin_amdgcn_wmma_f32_16x16x32_bf16(false, pa, false, vb,
                                                      (short)0, o[tt],
                                                      false, false);
    }
    __syncthreads();
  }

  // ---- finalize: out_tok += O / l  (zeros where row had no valid keys) -----
#pragma unroll
  for (int tt = 0; tt < 4; ++tt) {
#pragma unroll
    for (int r = 0; r < 8; ++r) {
      int row = g0 + r + 8 * half;
      int col = tt * 16 + lr;
      float val = (lst[r] > 0.0f) ? o[tt][r] / lst[r] : 0.0f;
      long long off = (long long)z * 4096 + (long long)row * 64 + col;
      out_tok[off] += val;
    }
  }
}

// ---------------------------------------------------------------------------
// out_x[n, h*64+d] = sum_g w[n,h,g] * out_tok[bi[n],h,g,d]   (bf16 out)
// ---------------------------------------------------------------------------
__global__ __launch_bounds__(512)
void out_x_kernel(const float* __restrict__ w, const float* __restrict__ out_tok,
                  const int* __restrict__ bi, bf16_t* __restrict__ oxb) {
  __shared__ float wrow[512];
  int n = blockIdx.x;
  int t = threadIdx.x;
  wrow[t] = w[(long long)n * 512 + t];
  __syncthreads();
  int b = bi[n];
  int h = t >> 6, d = t & 63;
  const float* ot = out_tok + ((long long)(b * 8 + h)) * 4096 + d;
  const float* wr = wrow + h * 64;
  float acc = 0.0f;
#pragma unroll 8
  for (int g = 0; g < 64; ++g) acc += wr[g] * ot[g * 64];
  oxb[(long long)n * 512 + t] = (bf16_t)acc;
}

// ---------------------------------------------------------------------------
// Host orchestration
// ---------------------------------------------------------------------------
static inline char* ws_take(char*& p, size_t bytes) {
  char* r = p;
  p += (bytes + 255) & ~(size_t)255;
  return r;
}

extern "C" void kernel_launch(void* const* d_in, const int* in_sizes, int n_in,
                              void* d_out, int out_size, void* d_ws, size_t ws_size,
                              hipStream_t stream) {
  const float* x      = (const float*)d_in[0];
  const int*   bi     = (const int*)  d_in[1];
  const float* son    = (const float*)d_in[2];
  const int*   sbi    = (const int*)  d_in[3];
  const float* temp   = (const float*)d_in[4];
  const float* W_fx   = (const float*)d_in[5];
  const float* b_fx   = (const float*)d_in[6];
  const float* W_x    = (const float*)d_in[7];
  const float* b_x    = (const float*)d_in[8];
  const float* W_sl   = (const float*)d_in[9];
  const float* b_sl   = (const float*)d_in[10];
  const float* Wq     = (const float*)d_in[11];
  const float* bq     = (const float*)d_in[12];
  const float* Wk     = (const float*)d_in[13];
  const float* bk     = (const float*)d_in[14];
  const float* Wv     = (const float*)d_in[15];
  const float* bv     = (const float*)d_in[16];
  const float* Wck    = (const float*)d_in[17];
  const float* bck    = (const float*)d_in[18];
  const float* Wcv    = (const float*)d_in[19];
  const float* bcv    = (const float*)d_in[20];
  const float* W_out  = (const float*)d_in[21];
  float* out = (float*)d_out;

  // ---- workspace carve-out -------------------------------------------------
  char* p = (char*)d_ws;
  bf16_t* xb     = (bf16_t*)ws_take(p, (size_t)N_TOK * 512 * 2);
  bf16_t* fxb    = (bf16_t*)ws_take(p, (size_t)N_TOK * 512 * 2);
  bf16_t* xmb    = (bf16_t*)ws_take(p, (size_t)N_TOK * 512 * 2);
  float*  logits = (float*) ws_take(p, (size_t)N_TOK * 512 * 4);   // -> w f32
  bf16_t* wbb    = (bf16_t*)ws_take(p, (size_t)N_TOK * 512 * 2);
  float*  tokens = (float*) ws_take(p, (size_t)BATCH * 8 * 4096 * 4);
  float*  norms  = (float*) ws_take(p, (size_t)BATCH * 8 * 64 * 4);
  bf16_t* tokb   = (bf16_t*)ws_take(p, (size_t)BATCH * 8 * 4096 * 2);
  bf16_t* qb     = (bf16_t*)ws_take(p, (size_t)BATCH * 8 * 4096 * 2);
  bf16_t* kb     = (bf16_t*)ws_take(p, (size_t)BATCH * 8 * 4096 * 2);
  bf16_t* vb     = (bf16_t*)ws_take(p, (size_t)BATCH * 8 * 4096 * 2);
  float*  attn   = (float*) ws_take(p, (size_t)BATCH * 8 * 4096 * 4);
  bf16_t* attnb  = (bf16_t*)ws_take(p, (size_t)BATCH * 8 * 4096 * 2);
  float*  otok   = (float*) ws_take(p, (size_t)BATCH * 8 * 4096 * 4);
  bf16_t* otb    = (bf16_t*)ws_take(p, (size_t)BATCH * 8 * 4096 * 2);
  bf16_t* sonb   = (bf16_t*)ws_take(p, (size_t)MSON * 512 * 2);
  bf16_t* ksb    = (bf16_t*)ws_take(p, (size_t)MSON * 512 * 2);
  bf16_t* vsb    = (bf16_t*)ws_take(p, (size_t)MSON * 512 * 2);
  bf16_t* oxb    = (bf16_t*)ws_take(p, (size_t)N_TOK * 512 * 2);
  bf16_t* wfxb   = (bf16_t*)ws_take(p, (size_t)512 * 512 * 2);
  bf16_t* wxb    = (bf16_t*)ws_take(p, (size_t)512 * 512 * 2);
  bf16_t* woutb  = (bf16_t*)ws_take(p, (size_t)512 * 512 * 2);
  bf16_t* wslb   = (bf16_t*)ws_take(p, (size_t)64 * 64 * 2);
  bf16_t* wqb    = (bf16_t*)ws_take(p, (size_t)64 * 64 * 2);
  bf16_t* wkb    = (bf16_t*)ws_take(p, (size_t)64 * 64 * 2);
  bf16_t* wvb    = (bf16_t*)ws_take(p, (size_t)64 * 64 * 2);
  bf16_t* wckb   = (bf16_t*)ws_take(p, (size_t)64 * 64 * 2);
  bf16_t* wcvb   = (bf16_t*)ws_take(p, (size_t)64 * 64 * 2);

  // ---- conversions to bf16 -------------------------------------------------
  f32_to_bf16<<<2048, 256, 0, stream>>>(x,     xb,    (long long)N_TOK * 512);
  f32_to_bf16<<<512,  256, 0, stream>>>(son,   sonb,  (long long)MSON * 512);
  f32_to_bf16<<<512,  256, 0, stream>>>(W_fx,  wfxb,  512 * 512);
  f32_to_bf16<<<512,  256, 0, stream>>>(W_x,   wxb,   512 * 512);
  f32_to_bf16<<<512,  256, 0, stream>>>(W_out, woutb, 512 * 512);
  f32_to_bf16<<<16,   256, 0, stream>>>(W_sl,  wslb,  64 * 64);
  f32_to_bf16<<<16,   256, 0, stream>>>(Wq,    wqb,   64 * 64);
  f32_to_bf16<<<16,   256, 0, stream>>>(Wk,    wkb,   64 * 64);
  f32_to_bf16<<<16,   256, 0, stream>>>(Wv,    wvb,   64 * 64);
  f32_to_bf16<<<16,   256, 0, stream>>>(Wck,   wckb,  64 * 64);
  f32_to_bf16<<<16,   256, 0, stream>>>(Wcv,   wcvb,  64 * 64);

  auto gemm = [&](const bf16_t* A, long long lda, int tA,
                  const bf16_t* B_, long long ldb, int tB,
                  const float* bias, float alpha,
                  float* Cf, bf16_t* Cb, long long ldc,
                  int M_, int Nn, int K_, int Z, int zdiv,
                  long long aO1, long long aO2, long long bO1, long long bO2,
                  long long cO1, long long cO2,
                  const int* kmask, int maskDiv) {
    dim3 grid(M_ / 64, Nn / 16, Z);
    if (kmask)
      gemm_wmma<1, 0, 1><<<grid, 32, 0, stream>>>(A, lda, B_, ldb, bias, alpha,
          Cf, Cb, ldc, K_, zdiv, aO1, aO2, bO1, bO2, cO1, cO2, kmask, maskDiv);
    else if (tB)
      gemm_wmma<0, 1, 0><<<grid, 32, 0, stream>>>(A, lda, B_, ldb, bias, alpha,
          Cf, Cb, ldc, K_, zdiv, aO1, aO2, bO1, bO2, cO1, cO2, nullptr, 1);
    else
      gemm_wmma<0, 0, 0><<<grid, 32, 0, stream>>>(A, lda, B_, ldb, bias, alpha,
          Cf, Cb, ldc, K_, zdiv, aO1, aO2, bO1, bO2, cO1, cO2, nullptr, 1);
    (void)tA;
  };

  // ---- fx = x@W_fx+b_fx (bf16), xm = x@W_x+b_x (bf16) ----------------------
  gemm(xb, 512, 0, wfxb, 512, 0, b_fx, 1.0f, nullptr, fxb, 512,
       N_TOK, 512, 512, 1, 1, 0,0, 0,0, 0,0, nullptr, 1);
  gemm(xb, 512, 0, wxb, 512, 0, b_x, 1.0f, nullptr, xmb, 512,
       N_TOK, 512, 512, 1, 1, 0,0, 0,0, 0,0, nullptr, 1);

  // ---- logits[n,h,g] = xm_h @ W_slice + b_slice (z = head) -----------------
  gemm(xmb, 512, 0, wslb, 64, 0, b_sl, 1.0f, logits, nullptr, 512,
       N_TOK, 64, 64, HEADS, 1, 64,0, 0,0, 64,0, nullptr, 1);

  // ---- w = softmax(logits / temp[h]) -> f32 in place + bf16 ----------------
  softmax_rows64<<<(N_TOK * HEADS) / 8, 256, 0, stream>>>(logits, wbb,
                                                          N_TOK * HEADS,
                                                          temp, HEADS);

  // ---- norms + masked token aggregation GEMM -------------------------------
  zero_f32<<<16, 256, 0, stream>>>(norms, BATCH * HEADS * GSLOT);
  {
    dim3 g(BATCH * HEADS, 16);
    norms_partial<<<g, 64, 0, stream>>>(logits, bi, norms, N_TOK / 16);
  }
  // tokens[b,h,g,d] = sum_n mask_b(n)*w[n,h,g]*fx[n,h,d]  (z=b*H+h, zdiv=H)
  gemm(wbb, 512, 1, fxb, 512, 0, nullptr, 1.0f, tokens, nullptr, 64,
       GSLOT, DHEAD, N_TOK, BATCH * HEADS, HEADS,
       0, 64, 0, 64, (long long)HEADS * 4096, 4096, bi, HEADS);
  tokens_finalize<<<(BATCH * HEADS * 4096) / 256, 256, 0, stream>>>(
      tokens, norms, tokb, BATCH * HEADS * 4096);

  // ---- q,k,v = tokens @ W* + b* (per (b,h)) --------------------------------
  gemm(tokb, 64, 0, wqb, 64, 0, bq, 1.0f, nullptr, qb, 64,
       GSLOT, DHEAD, DHEAD, BATCH * HEADS, 1, 4096,0, 0,0, 4096,0, nullptr, 1);
  gemm(tokb, 64, 0, wkb, 64, 0, bk, 1.0f, nullptr, kb, 64,
       GSLOT, DHEAD, DHEAD, BATCH * HEADS, 1, 4096,0, 0,0, 4096,0, nullptr, 1);
  gemm(tokb, 64, 0, wvb, 64, 0, bv, 1.0f, nullptr, vb, 64,
       GSLOT, DHEAD, DHEAD, BATCH * HEADS, 1, 4096,0, 0,0, 4096,0, nullptr, 1);

  // ---- attn = softmax(q @ k^T * SCALE) -------------------------------------
  gemm(qb, 64, 0, kb, 64, 1, nullptr, SCALE, attn, nullptr, 64,
       GSLOT, GSLOT, DHEAD, BATCH * HEADS, 1, 4096,0, 4096,0, 4096,0, nullptr, 1);
  softmax_rows64<<<(BATCH * HEADS * GSLOT) / 8, 256, 0, stream>>>(
      attn, attnb, BATCH * HEADS * GSLOT, nullptr, 1);

  // ---- out_tok = attn @ v (f32 + bf16 copy for the flash Q operand) --------
  gemm(attnb, 64, 0, vb, 64, 0, nullptr, 1.0f, otok, otb, 64,
       GSLOT, DHEAD, GSLOT, BATCH * HEADS, 1, 4096,0, 4096,0, 4096,0, nullptr, 1);

  // ---- ks, vs = sonata_h @ Wck/Wcv + b -------------------------------------
  gemm(sonb, 512, 0, wckb, 64, 0, bck, 1.0f, nullptr, ksb, 512,
       MSON, DHEAD, DHEAD, HEADS, 1, 64,0, 0,0, 64,0, nullptr, 1);
  gemm(sonb, 512, 0, wcvb, 64, 0, bcv, 1.0f, nullptr, vsb, 512,
       MSON, DHEAD, DHEAD, HEADS, 1, 64,0, 0,0, 64,0, nullptr, 1);

  // ---- fused context attention: out_tok += softmax(masked scores) @ vs -----
  {
    dim3 g(GSLOT / 16, BATCH * HEADS);
    ctx_attn<<<g, 32, 0, stream>>>(otb, ksb, vsb, sbi, otok);
  }

  // ---- out_x[n,h,d] = sum_g w[n,h,g] * out_tok[bi[n],h,g,d] ----------------
  out_x_kernel<<<N_TOK, 512, 0, stream>>>(logits, otok, bi, oxb);

  // ---- final: out = out_x @ W_out (f32 output) -----------------------------
  gemm(oxb, 512, 0, woutb, 512, 0, nullptr, 1.0f, out, nullptr, 512,
       N_TOK, 512, 512, 1, 1, 0,0, 0,0, 0,0, nullptr, 1);
}
